// SwinT_75325136437265
// MI455X (gfx1250) — matrix operand order repaired and checked by reading, not compiled, not run on hardware
//
#include <hip/hip_runtime.h>
#include <hip/hip_bf16.h>

// ---------------------------------------------------------------------------
// Swin-T forward on gfx1250: WMMA f16 GEMMs fed by async global->LDS staging
// (double-buffered, ASYNCcnt-tracked), fused window gather/scatter + bias /
// GELU / residual epilogues, single-wave in-LDS window attention.
// ---------------------------------------------------------------------------

typedef __attribute__((ext_vector_type(16))) _Float16 v16h;
typedef __attribute__((ext_vector_type(8)))  _Float16 v8h;
typedef __attribute__((ext_vector_type(8)))  float    v8f;
typedef __attribute__((ext_vector_type(16))) float    v16f;

// ---------------- async global -> LDS (CDNA5 VFLAT async, ASYNCcnt) --------
__device__ __forceinline__ void async_b128(void* lds_dst, const void* gsrc) {
  // GV mode: 64-bit vaddr, saddr = off. LDS dest VGPR = wave-relative LDS
  // byte address (low 32 bits of the flat shared-aperture pointer).
  asm volatile("global_load_async_to_lds_b128 %0, %1, off"
               :
               : "v"((unsigned)(unsigned long)lds_dst),
                 "v"((unsigned long)gsrc)
               : "memory");
}

// ---------------- WMMA fragment helpers (wave32, 16x16x32 f16) -------------
// A (MxK) f16 LDS row-major: lane m=l&15, half h=l>>4; VGPR0-3 K=h*8+0..7,
// VGPR4-7 K=16+h*8+0..7.
__device__ __forceinline__ v16h frag_a(const _Float16* lds, int row0, int ld,
                                       int k0, int lane) {
  const int m  = row0 + (lane & 15);
  const int kh = (lane >> 4) * 8;
  const _Float16* p = lds + m * ld + k0 + kh;
  v8h lo = *(const v8h*)(p);
  v8h hi = *(const v8h*)(p + 16);
  return __builtin_shufflevector(lo, hi, 0, 1, 2, 3, 4, 5, 6, 7,
                                 8, 9, 10, 11, 12, 13, 14, 15);
}

// B (KxN) stored transposed [n][k] f16: half h=l>>4; VGPR0-3 K=h*8+0..7,
// VGPR4-7 K=16+h*8+0..7.
__device__ __forceinline__ v16h frag_b(const _Float16* ldsT, int col0, int ld,
                                       int k0, int lane) {
  const int n = col0 + (lane & 15);
  const _Float16* p = ldsT + n * ld + k0 + (lane >> 4) * 8;
  v8h lo = *(const v8h*)(p);
  v8h hi = *(const v8h*)(p + 16);
  return __builtin_shufflevector(lo, hi, 0, 1, 2, 3, 4, 5, 6, 7,
                                 8, 9, 10, 11, 12, 13, 14, 15);
}

// f32-staged variants (async path): convert to f16 at fragment build.
__device__ __forceinline__ v16h frag_a32(const float* lds, int row0, int ld,
                                         int k0, int lane) {
  const int m  = row0 + (lane & 15);
  const int kh = (lane >> 4) * 8;
  const float* p = lds + m * ld + k0 + kh;
  v8f lo = *(const v8f*)(p);
  v8f hi = *(const v8f*)(p + 16);
  v16f f = __builtin_shufflevector(lo, hi, 0, 1, 2, 3, 4, 5, 6, 7,
                                   8, 9, 10, 11, 12, 13, 14, 15);
  return __builtin_convertvector(f, v16h);
}

__device__ __forceinline__ v16h frag_b32(const float* ldsT, int col0, int ld,
                                         int k0, int lane) {
  const int n = col0 + (lane & 15);
  const float* p = ldsT + n * ld + k0 + (lane >> 4) * 8;
  v8f lo = *(const v8f*)(p);
  v8f hi = *(const v8f*)(p + 16);
  v16f f = __builtin_shufflevector(lo, hi, 0, 1, 2, 3, 4, 5, 6, 7,
                                   8, 9, 10, 11, 12, 13, 14, 15);
  return __builtin_convertvector(f, v16h);
}

__device__ __forceinline__ v8f wmma16(v16h a, v16h b, v8f c) {
  return __builtin_amdgcn_wmma_f32_16x16x32_f16(false, a, false, b,
                                                (short)0, c, false, false);
}

__device__ __forceinline__ float gelu_exact(float v) {
  return 0.5f * v * (1.0f + erff(v * 0.7071067811865476f));
}

// Window-token row -> element offset in (B,H,W,C) image layout, applying the
// roll(-s) gather (same map serves the roll(+s) scatter on the output side).
__device__ __forceinline__ long win_img_index(int m, int H, int W, int C,
                                              int s) {
  const int q   = m % 49;
  const int win = m / 49;
  const int nw  = W / 7;
  const int nh  = H / 7;
  const int wj  = win % nw;
  const int t   = win / nw;
  const int wi  = t % nh;
  const int b   = t / nh;
  int r = wi * 7 + q / 7 + s; if (r >= H) r -= H;
  int c = wj * 7 + q % 7 + s; if (c >= W) c -= W;
  return ((long)(b * H + r) * W + c) * C;
}

// ---------------- Generic fused GEMM:  out = epi(A @ Wt^T) ------------------
// MODE 0: plain / plain.  MODE 1: A gathered via window partition (qkv in).
// MODE 2: out scattered via window partition (proj out).
// flags: 1=bias, 2=gelu, 4=residual add at out index.
// Interior 64x64 tiles stream both operands with double-buffered async
// global->LDS b128 copies; edge tiles use a zero-filled synchronous path.
template <int MODE>
__global__ __launch_bounds__(128) void gemm_wmma(
    const float* __restrict__ A, const float* __restrict__ Wt,
    const float* __restrict__ bias, const float* __restrict__ resid,
    float* __restrict__ out, int M, int N, int K, int H, int W, int shift,
    int flags) {
  __shared__ __align__(64) float As[2][64 * 32];
  __shared__ __align__(64) float Bs[2][64 * 32];

  const int ntN  = (N + 63) >> 6;
  const int tm   = (blockIdx.x / ntN) * 64;
  const int tn   = (blockIdx.x % ntN) * 64;
  const int tid  = threadIdx.x;
  const int lane = tid & 31;
  const int wave = tid >> 5;

  v8f acc[4] = {v8f{}, v8f{}, v8f{}, v8f{}};

  // K is a multiple of 32 for every GEMM in this net; interior predicate is
  // uniform per block.
  const bool interior = (tm + 64 <= M) && (tn + 64 <= N);

  auto stage_async = [&](int buf, int k0) {
#pragma unroll
    for (int j = 0; j < 4; ++j) {
      const int cid = tid + j * 128;  // 0..511 b128 chunks per operand
      const int r   = cid >> 3;       // tile row 0..63
      const int c   = (cid & 7) * 4;  // col 0,4,...,28
      long abase;
      if (MODE == 1)
        abase = win_img_index(tm + r, H, W, K, shift);
      else
        abase = (long)(tm + r) * K;
      async_b128(&As[buf][r * 32 + c], A + abase + k0 + c);
      async_b128(&Bs[buf][r * 32 + c], Wt + (long)(tn + r) * K + k0 + c);
    }
  };

  auto stage_sync = [&](int k0) {
    const int r   = tid >> 1;
    const int cb  = (tid & 1) * 16;
    const int row = tm + r;
    long abase = -1;
    if (row < M) {
      if (MODE == 1)
        abase = win_img_index(row, H, W, K, shift);
      else
        abase = (long)row * K;
      if (k0 + 32 < K) __builtin_prefetch(A + abase + k0 + 32, 0, 1);
    }
#pragma unroll
    for (int i = 0; i < 16; ++i) {
      const int k = k0 + cb + i;
      As[0][r * 32 + cb + i] = (abase >= 0 && k < K) ? A[abase + k] : 0.0f;
    }
    const int n = tn + r;
#pragma unroll
    for (int i = 0; i < 16; ++i) {
      const int k = k0 + cb + i;
      Bs[0][r * 32 + cb + i] =
          (n < N && k < K) ? Wt[(long)n * K + k] : 0.0f;
    }
  };

  auto compute = [&](int buf) {
    v16h af = frag_a32(As[buf], wave * 16, 32, 0, lane);
#pragma unroll
    for (int t = 0; t < 4; ++t) {
      v16h bf = frag_b32(Bs[buf], t * 16, 32, 0, lane);
      acc[t]  = wmma16(af, bf, acc[t]);
    }
  };

  if (interior) {
    stage_async(0, 0);
    int buf = 0;
    for (int k0 = 0; k0 < K; k0 += 32) {
      if (k0 + 32 < K) {
        stage_async(buf ^ 1, k0 + 32);
        // 8 outstanding = next tile's copies; async loads complete in order,
        // so <=8 pending implies the current tile has landed in LDS.
        asm volatile("s_wait_asynccnt 0x8" ::: "memory");
      } else {
        asm volatile("s_wait_asynccnt 0x0" ::: "memory");
      }
      __syncthreads();
      compute(buf);
      __syncthreads();
      buf ^= 1;
    }
  } else {
    for (int k0 = 0; k0 < K; k0 += 32) {
      stage_sync(k0);
      __syncthreads();
      compute(0);
      __syncthreads();
    }
  }

  // ---- epilogue: bias / gelu / residual, plain or window-scatter store
  const int hl   = (lane >> 4) * 8;
  const int col0 = tn + (lane & 15);
#pragma unroll
  for (int r = 0; r < 8; ++r) {
    const int row = tm + wave * 16 + r + hl;
    if (row >= M) continue;
    long obase;
    if (MODE == 2)
      obase = win_img_index(row, H, W, N, shift);
    else
      obase = (long)row * N;
#pragma unroll
    for (int t = 0; t < 4; ++t) {
      const int col = col0 + t * 16;
      if (col >= N) continue;
      float v = acc[t][r];
      if (flags & 1) v += bias[col];
      if (flags & 2) v = gelu_exact(v);
      const long oidx = obase + col;
      if (flags & 4) v += resid[oidx];
      out[oidx] = v;
    }
  }
}

// ---------------- Window attention: one wave per (window, head) -------------
// head_dim == 32 everywhere. 49 tokens padded to 64. Entirely in LDS.
__global__ __launch_bounds__(32) void win_attn(
    const float* __restrict__ qkv, const float* __restrict__ rpb,
    float* __restrict__ o, int heads, int C, int H, int W, int shift) {
  const int win  = blockIdx.x;
  const int h    = blockIdx.y;
  const int lane = threadIdx.x;

  __shared__ __align__(16) _Float16 Qs[64 * 32];
  __shared__ __align__(16) _Float16 Ks[64 * 32];
  __shared__ __align__(16) _Float16 Vt[32 * 64];  // transposed: [dim][tok]
  __shared__ __align__(16) float    Sm[64 * 64];
  __shared__ __align__(16) _Float16 Pm[64 * 64];

  const int   C3    = 3 * C;
  const float scale = 0.17677669529663687f;  // 1/sqrt(32)

  for (int idx = lane; idx < 64 * 32; idx += 32) {
    const int r = idx >> 5, d = idx & 31;
    float q = 0.0f, k = 0.0f, v = 0.0f;
    if (r < 49) {
      const long base = (long)(win * 49 + r) * C3 + h * 32 + d;
      q = qkv[base];
      k = qkv[base + C];
      v = qkv[base + 2 * C];
    }
    Qs[idx]        = (_Float16)(q * scale);
    Ks[idx]        = (_Float16)k;
    Vt[d * 64 + r] = (_Float16)v;
  }
  __syncthreads();

  // S = Q K^T  (64x64, K=32: single WMMA per 16x16 tile)
#pragma unroll
  for (int ti = 0; ti < 4; ++ti) {
    v16h af = frag_a(Qs, ti * 16, 32, 0, lane);
#pragma unroll
    for (int tj = 0; tj < 4; ++tj) {
      v16h bf = frag_b(Ks, tj * 16, 32, 0, lane);
      v8f  c  = {};
      c = wmma16(af, bf, c);
#pragma unroll
      for (int r = 0; r < 8; ++r)
        Sm[(ti * 16 + r + ((lane >> 4) << 3)) * 64 + tj * 16 + (lane & 15)] =
            c[r];
    }
  }
  __syncthreads();

  // softmax rows (+relative position bias, +shift mask), lane owns rows.
  const int nw = W / 7;
  const int wj = win % nw;
  const int t2 = win / nw;
  const int nh = H / 7;
  const int wi = t2 % nh;
  for (int q = lane; q < 64; q += 32) {
    if (q < 49) {
      const int qi = q / 7, qj = q % 7;
      int cq = 0;
      if (shift) {
        const int rr = wi * 7 + qi, cc = wj * 7 + qj;
        const int ri = rr < H - 7 ? 0 : (rr < H - shift ? 1 : 2);
        const int ci = cc < W - 7 ? 0 : (cc < W - shift ? 1 : 2);
        cq = ri * 3 + ci;
      }
      float mx = -1e30f;
      for (int k = 0; k < 49; ++k) {
        const int ki = k / 7, kj = k % 7;
        float s = Sm[q * 64 + k] +
                  rpb[((qi - ki + 6) * 13 + (qj - kj + 6)) * heads + h];
        if (shift) {
          const int rr = wi * 7 + ki, cc = wj * 7 + kj;
          const int ri = rr < H - 7 ? 0 : (rr < H - shift ? 1 : 2);
          const int ci = cc < W - 7 ? 0 : (cc < W - shift ? 1 : 2);
          if (ri * 3 + ci != cq) s -= 100.0f;
        }
        Sm[q * 64 + k] = s;
        mx = fmaxf(mx, s);
      }
      float sum = 0.0f;
      for (int k = 0; k < 49; ++k) {
        const float e = __expf(Sm[q * 64 + k] - mx);
        Sm[q * 64 + k] = e;
        sum += e;
      }
      const float inv = 1.0f / sum;
      for (int k = 0; k < 49; ++k)
        Pm[q * 64 + k] = (_Float16)(Sm[q * 64 + k] * inv);
      for (int k = 49; k < 64; ++k) Pm[q * 64 + k] = (_Float16)0.0f;
    } else {
      for (int k = 0; k < 64; ++k) Pm[q * 64 + k] = (_Float16)0.0f;
    }
  }
  __syncthreads();

  // O = P V  (64x32, K=64: two chained k=32 WMMAs per tile)
#pragma unroll
  for (int ti = 0; ti < 4; ++ti) {
#pragma unroll
    for (int tj = 0; tj < 2; ++tj) {
      v8f c = {};
#pragma unroll
      for (int kk = 0; kk < 2; ++kk) {
        v16h af = frag_a(Pm, ti * 16, 64, kk * 32, lane);
        v16h bf = frag_b(Vt, tj * 16, 64, kk * 32, lane);
        c = wmma16(af, bf, c);
      }
#pragma unroll
      for (int r = 0; r < 8; ++r) {
        const int row = ti * 16 + r + ((lane >> 4) << 3);
        if (row < 49)
          o[(long)(win * 49 + row) * C + h * 32 + tj * 16 + (lane & 15)] =
              c[r];
      }
    }
  }
}

// ---------------- LayerNorm: one wave per token ----------------------------
__global__ __launch_bounds__(32) void layernorm(
    const float* __restrict__ x, const float* __restrict__ w,
    const float* __restrict__ b, float* __restrict__ out, int C) {
  const long tok  = blockIdx.x;
  const int  lane = threadIdx.x;
  const float* p  = x + tok * C;
  float s = 0.0f, s2 = 0.0f;
  for (int c = lane; c < C; c += 32) {
    const float v = p[c];
    s += v;
    s2 += v * v;
  }
#pragma unroll
  for (int off = 16; off; off >>= 1) {
    s  += __shfl_xor(s, off, 32);
    s2 += __shfl_xor(s2, off, 32);
  }
  const float m   = s / C;
  const float var = s2 / C - m * m;
  const float rr  = rsqrtf(var + 1e-5f);
  float* q = out + tok * C;
  for (int c = lane; c < C; c += 32) q[c] = (p[c] - m) * rr * w[c] + b[c];
}

// ---------------- Patch embed: conv4x4/s4 + input normalization ------------
__global__ void patch_embed(const float* __restrict__ x,
                            const float* __restrict__ cw,
                            const float* __restrict__ cb,
                            float* __restrict__ out, int Bn) {
  const long i     = (long)blockIdx.x * blockDim.x + threadIdx.x;
  const long total = (long)Bn * 56 * 56 * 96;
  if (i >= total) return;
  const int co = i % 96;
  long t = i / 96;
  const int ow = t % 56; t /= 56;
  const int oh = t % 56;
  const int b  = (int)(t / 56);
  const float mean[3] = {0.485f, 0.456f, 0.406f};
  const float stdv[3] = {0.229f, 0.224f, 0.225f};
  float acc = cb[co];
#pragma unroll
  for (int ci = 0; ci < 3; ++ci) {
    const float mu = mean[ci], sd = 1.0f / stdv[ci];
#pragma unroll
    for (int kh = 0; kh < 4; ++kh)
#pragma unroll
      for (int kw = 0; kw < 4; ++kw) {
        float xv = x[(((long)b * 3 + ci) * 224 + oh * 4 + kh) * 224 +
                     ow * 4 + kw];
        xv = ((xv + 1.0f) * 0.5f - mu) * sd;
        acc += xv * cw[((co * 3 + ci) * 4 + kh) * 4 + kw];
      }
  }
  out[i] = acc;
}

// ---------------- Patch merging gather: (B,H,W,C) -> (B,H/2,W/2,4C) --------
__global__ void merge_gather(const float* __restrict__ y,
                             float* __restrict__ out, int Bn, int H, int W,
                             int C) {
  const long i     = (long)blockIdx.x * blockDim.x + threadIdx.x;
  const long total = (long)Bn * (H / 2) * (W / 2) * 4 * C;
  if (i >= total) return;
  const int c4 = i % (4 * C);
  long t = i / (4 * C);
  const int j  = t % (W / 2); t /= (W / 2);
  const int ii = t % (H / 2);
  const int b  = (int)(t / (H / 2));
  const int part = c4 / C, c = c4 % C;
  const int r  = 2 * ii + (part & 1);
  const int cc = 2 * j + (part >> 1);
  out[i] = y[((long)(b * H + r) * W + cc) * C + c];
}

// ---------------------------------------------------------------------------
// Host orchestration.
// Input flattening assumed: d_in[0] = x, then params with sorted keys:
//   blocks[0..11] x {m1b,m1w,m2b,m2w,n1b,n1w,n2b,n2w,projb,projw,qkvb,qkvw,rpb}
//   conv_b, conv_w, ln0_b, ln0_w, merges[0..2] x {nb,nw,rw}
// ---------------------------------------------------------------------------
extern "C" void kernel_launch(void* const* d_in, const int* in_sizes, int n_in,
                              void* d_out, int out_size, void* d_ws,
                              size_t ws_size, hipStream_t stream) {
  (void)in_sizes; (void)n_in; (void)out_size; (void)ws_size;
  auto P = [&](int i) { return (const float*)d_in[i]; };
  const float* x = P(0);
  const int IDX_CONV_B = 1 + 12 * 13 + 0;
  const int IDX_CONV_W = 1 + 12 * 13 + 1;
  const int IDX_LN0_B  = 1 + 12 * 13 + 2;
  const int IDX_LN0_W  = 1 + 12 * 13 + 3;
  const int IDX_MERGE0 = 1 + 12 * 13 + 4;

  // Workspace layout (floats).
  float* ws    = (float*)d_ws;
  float* y     = ws;               // residual stream     9,633,792
  float* lnb   = y     +  9633792; // LN output           9,633,792
  float* qkvb_ = lnb   +  9633792; // qkv                28,901,376
  float* attn  = qkvb_ + 28901376; // attn out / temp     9,633,792
  float* hbuf  = attn  +  9633792; // mlp hidden / LN    38,535,168
  float* mrg   = hbuf  + 38535168; // merge concat        9,633,792

  static const int DIMS[4]  = {96, 192, 384, 768};
  static const int HEADS[4] = {3, 6, 12, 24};
  static const int DEPTH[4] = {2, 2, 6, 2};
  const int B = 32;

  // Patch embed -> attn (temp), then ln0 -> y.
  {
    const long total = (long)B * 56 * 56 * 96;
    patch_embed<<<(unsigned)((total + 255) / 256), 256, 0, stream>>>(
        x, P(IDX_CONV_W), P(IDX_CONV_B), attn, B);
    layernorm<<<(unsigned)(B * 56 * 56), 32, 0, stream>>>(
        attn, P(IDX_LN0_W), P(IDX_LN0_B), y, 96);
  }

  int H = 56, W = 56, bi = 0;
  for (int st = 0; st < 4; ++st) {
    const int C = DIMS[st], heads = HEADS[st];
    for (int d = 0; d < DEPTH[st]; ++d, ++bi) {
      const int s  = (d % 2 == 1 && H > 7) ? 3 : 0;
      const int pb = 1 + bi * 13;
      const float *m1b = P(pb + 0), *m1w = P(pb + 1), *m2b = P(pb + 2),
                  *m2w = P(pb + 3), *n1b = P(pb + 4), *n1w = P(pb + 5),
                  *n2b = P(pb + 6), *n2w = P(pb + 7), *projb = P(pb + 8),
                  *projw = P(pb + 9), *qkvbias = P(pb + 10),
                  *qkvw = P(pb + 11), *rpb = P(pb + 12);
      const int M = B * H * W;

      // LN1
      layernorm<<<(unsigned)M, 32, 0, stream>>>(y, n1w, n1b, lnb, C);
      // QKV: gather shifted windows from lnb, fused bias.
      {
        const int N = 3 * C, K = C;
        const unsigned g = (unsigned)(((M + 63) / 64) * ((N + 63) / 64));
        gemm_wmma<1><<<g, 128, 0, stream>>>(lnb, qkvw, qkvbias, nullptr,
                                            qkvb_, M, N, K, H, W, s, 1);
      }
      // Attention per (window, head), entirely in LDS + WMMA.
      {
        dim3 g((unsigned)(B * (H / 7) * (W / 7)), (unsigned)heads);
        win_attn<<<g, 32, 0, stream>>>(qkvb_, rpb, attn, heads, C, H, W, s);
      }
      // Proj: scatter back to image layout (+roll), fused residual into y.
      {
        const unsigned g = (unsigned)(((M + 63) / 64) * ((C + 63) / 64));
        gemm_wmma<2><<<g, 128, 0, stream>>>(attn, projw, projb, y, y, M, C,
                                            C, H, W, s, 1 | 4);
      }
      // LN2
      layernorm<<<(unsigned)M, 32, 0, stream>>>(y, n2w, n2b, lnb, C);
      // MLP fc1 + exact GELU.
      {
        const int N = 4 * C;
        const unsigned g = (unsigned)(((M + 63) / 64) * ((N + 63) / 64));
        gemm_wmma<0><<<g, 128, 0, stream>>>(lnb, m1w, m1b, nullptr, hbuf, M,
                                            N, C, 0, 0, 0, 1 | 2);
      }
      // MLP fc2 + residual into y.
      {
        const unsigned g = (unsigned)(((M + 63) / 64) * ((C + 63) / 64));
        gemm_wmma<0><<<g, 128, 0, stream>>>(hbuf, m2w, m2b, y, y, M, C,
                                            4 * C, 0, 0, 0, 1 | 4);
      }
    }
    if (st < 3) {
      const int mb = IDX_MERGE0 + st * 3;
      const float *nb = P(mb + 0), *nw = P(mb + 1), *rw = P(mb + 2);
      const int M2 = B * (H / 2) * (W / 2);
      const long tot = (long)M2 * 4 * C;
      merge_gather<<<(unsigned)((tot + 255) / 256), 256, 0, stream>>>(
          y, mrg, B, H, W, C);
      layernorm<<<(unsigned)M2, 32, 0, stream>>>(mrg, nw, nb, hbuf, 4 * C);
      const int N = 2 * C, K = 4 * C;
      const unsigned g = (unsigned)(((M2 + 63) / 64) * ((N + 63) / 64));
      gemm_wmma<0><<<g, 128, 0, stream>>>(hbuf, rw, nullptr, nullptr, y, M2,
                                          N, K, 0, 0, 0, 0);
      H /= 2;
      W /= 2;
    }
  }

  hipMemcpyAsync(d_out, y, sizeof(float) * 32 * 7 * 7 * 768,
                 hipMemcpyDeviceToDevice, stream);
}
